// KLayerHeteroGAT_30133490549161
// MI455X (gfx1250) — matrix operand order repaired and checked by reading, not compiled, not run on hardware
//
#include <hip/hip_runtime.h>
#include <hip/hip_bf16.h>

typedef __attribute__((ext_vector_type(16))) _Float16 v16h;
typedef __attribute__((ext_vector_type(8)))  _Float16 v8h;
typedef __attribute__((ext_vector_type(8)))  float    v8f;

#define NNODES 50000
#define NEDGES 800000
#define DH     128

// ---- monotone float <-> uint key for atomicMax on signed floats ----
__device__ __forceinline__ unsigned fkey(float f) {
    unsigned u = __float_as_uint(f);
    return (u & 0x80000000u) ? ~u : (u | 0x80000000u);
}
__device__ __forceinline__ float funkey(unsigned k) {
    unsigned u = (k & 0x80000000u) ? (k & 0x7FFFFFFFu) : ~k;
    return __uint_as_float(u);
}

// ---- init all accumulators in one pass (grid covers N*128) ----
__global__ void init_kernel(float* agg, unsigned* emaxu, float* esum, int* deg, float* hg) {
    int i = blockIdx.x * 256 + threadIdx.x;
    if (i < NNODES * DH) agg[i] = 0.0f;
    if (i < 2 * NNODES) { emaxu[i] = 0u; esum[i] = 0.0f; }
    if (i < NNODES)     deg[i] = 0;
    if (i < DH)         hg[i] = 0.0f;
}

// ---- convert x (f32 row-major) -> f16 row-major ----
__global__ void cvt_x(const float* __restrict__ x, _Float16* __restrict__ xh) {
    int i = blockIdx.x * 256 + threadIdx.x;   // exact N*128
    xh[i] = (_Float16)x[i];
}

// ---- convert W0,W1 (f32 [K][N]) -> f16 transposed [rel][N][K] ----
__global__ void cvt_w(const float* __restrict__ W0, const float* __restrict__ W1,
                      _Float16* __restrict__ wt) {
    int i = blockIdx.x * 256 + threadIdx.x;   // exact 2*128*128
    int rel = i >> 14;
    int r   = i & 16383;
    int n   = r >> 7;
    int k   = r & 127;
    const float* W = rel ? W1 : W0;
    wt[i] = (_Float16)W[k * DH + n];
}

// ---- WMMA GEMM: wh[rel] = x @ W_rel + b_rel, one wave per 16x16 tile ----
__global__ void gemm_wmma(const _Float16* __restrict__ xh,
                          const _Float16* __restrict__ wt,   // [2][128 n][128 k] f16
                          const float* __restrict__ b0, const float* __restrict__ b1,
                          float* __restrict__ wh) {          // [2][N][128] f32
    const int wave = threadIdx.x >> 5;
    const int lane = threadIdx.x & 31;
    const int t    = blockIdx.x * 8 + wave;            // 0 .. 49999 (exact)
    const int tilesPerRel = (NNODES / 16) * 8;         // 25000
    const int rel  = t / tilesPerRel;
    const int r    = t - rel * tilesPerRel;
    const int row0 = (r >> 3) * 16;
    const int col0 = (r & 7) * 16;
    const int half   = lane >> 4;                      // 0|1
    const int lane16 = lane & 15;

    const _Float16* xrow = xh + (size_t)(row0 + lane16) * DH;
    const _Float16* wcol = wt + (size_t)rel * DH * DH + (size_t)(col0 + lane16) * DH;
    const int aoff = half * 8;    // A: K chunks [aoff, aoff+8) and [aoff+16, aoff+24)
    const int boff = half * 8;    // B: K chunks [boff, boff+8) and [boff+16, boff+24)

    v8f acc = {};
#pragma unroll
    for (int k0 = 0; k0 < DH; k0 += 32) {
        v8h alo = *(const v8h*)(xrow + k0 + aoff);
        v8h ahi = *(const v8h*)(xrow + k0 + aoff + 16);
        v8h blo = *(const v8h*)(wcol + k0 + boff);
        v8h bhi = *(const v8h*)(wcol + k0 + boff + 16);
        v16h A, B;
#pragma unroll
        for (int i = 0; i < 8; ++i) {
            A[i] = alo[i]; A[i + 8] = ahi[i];
            B[i] = blo[i]; B[i + 8] = bhi[i];
        }
        acc = __builtin_amdgcn_wmma_f32_16x16x32_f16(false, A, false, B,
                                                     (short)0, acc, false, false);
    }

    const float* bias = rel ? b1 : b0;
    const float  bn   = bias[col0 + lane16];
    float* out = wh + (size_t)rel * NNODES * DH;
#pragma unroll
    for (int g = 0; g < 8; ++g) {
        int m = g + half * 8;
        out[(size_t)(row0 + m) * DH + col0 + lane16] = acc[g] + bn;
    }
}

// ---- per-node attention logits: es = wh@a[:128], ed = wh@a[128:] (wave per node) ----
__global__ void esed_kernel(const float* __restrict__ wh,
                            const float* __restrict__ a0, const float* __restrict__ a1,
                            float* __restrict__ es, float* __restrict__ ed) {
    int gid  = blockIdx.x * 256 + threadIdx.x;
    int w    = gid >> 5;                  // 0 .. 2N-1 (exact)
    int lane = gid & 31;
    int rel  = w / NNODES;
    int v    = w - rel * NNODES;
    const float* row = wh + ((size_t)rel * NNODES + v) * DH;
    const float* a   = rel ? a1 : a0;
    float s = 0.0f, d = 0.0f;
#pragma unroll
    for (int i = 0; i < 4; ++i) {
        int c = lane + i * 32;
        float x = row[c];
        s += x * a[c];
        d += x * a[DH + c];
    }
    for (int o = 16; o; o >>= 1) {
        s += __shfl_down(s, o, 32);
        d += __shfl_down(d, o, 32);
    }
    if (lane == 0) {
        es[(size_t)rel * NNODES + v] = s;
        ed[(size_t)rel * NNODES + v] = d;
    }
}

// ---- edge pass 1: leaky_relu logit, store, segment max over dst ----
__global__ void edge_max(const int* __restrict__ src, const int* __restrict__ dst,
                         const float* __restrict__ es, const float* __restrict__ ed,
                         float* __restrict__ eval, unsigned* __restrict__ emaxu) {
    int i = blockIdx.x * 256 + threadIdx.x;   // exact E
    int s = src[i], d = dst[i];
    float v = es[s] + ed[d];
    v = v > 0.0f ? v : 0.01f * v;             // torch leaky_relu slope
    eval[i] = v;
    atomicMax(emaxu + d, fkey(v));
}

// ---- edge pass 2: exp(e - max), segment sum over dst ----
__global__ void edge_exp(const int* __restrict__ dst, float* __restrict__ eval,
                         const unsigned* __restrict__ emaxu, float* __restrict__ esum) {
    int i = blockIdx.x * 256 + threadIdx.x;   // exact E
    int d = dst[i];
    float ex = __expf(eval[i] - funkey(emaxu[d]));
    eval[i] = ex;
    atomicAdd(esum + d, ex);
}

// ---- edge pass 3: agg[dst] += alpha * wh[src]  (wave per edge, float4 per lane) ----
__global__ void edge_scatter(const int* __restrict__ src, const int* __restrict__ dst,
                             const float* __restrict__ eval, const float* __restrict__ esum,
                             const float* __restrict__ wh, float* __restrict__ agg) {
    int gid  = blockIdx.x * 256 + threadIdx.x;
    int e    = gid >> 5;                      // exact E
    int lane = gid & 31;
    int s = src[e], d = dst[e];
    float alpha = eval[e] / esum[d];
    const float4 v = ((const float4*)(wh + (size_t)s * DH))[lane];
    float* o = agg + (size_t)d * DH + lane * 4;
    atomicAdd(o + 0, alpha * v.x);
    atomicAdd(o + 1, alpha * v.y);
    atomicAdd(o + 2, alpha * v.z);
    atomicAdd(o + 3, alpha * v.w);
}

// ---- out-degree histogram over src (readout: mean_nodes(sum(h[src])) == (1/N) sum deg*h) ----
__global__ void deg_kernel(const int* __restrict__ src, int* __restrict__ deg) {
    int i = blockIdx.x * 256 + threadIdx.x;   // exact E
    atomicAdd(deg + src[i], 1);
}

// ---- elu + degree-weighted channel reduction -> hg[128] ----
__global__ void readout(const float* __restrict__ agg, const int* __restrict__ deg,
                        float* __restrict__ hg) {
    int c  = threadIdx.x;                     // 0..127
    int v0 = blockIdx.x * 250;                // 200 blocks * 250 = N exact
    float acc = 0.0f;
    for (int v = v0; v < v0 + 250; ++v) {
        float a = agg[(size_t)v * DH + c];
        float h = a > 0.0f ? a : (__expf(a) - 1.0f);   // elu
        acc += (float)deg[v] * h;
    }
    atomicAdd(hg + c, acc);
}

// ---- final classifier: sigmoid(hg/N @ Wout + bout) ----
__global__ void final_kernel(const float* __restrict__ hg, const float* __restrict__ Wout,
                             const float* __restrict__ bout, float* __restrict__ out) {
    int j = threadIdx.x;
    if (j < 8) {
        float s = bout[j];
        const float invN = 1.0f / (float)NNODES;
        for (int c = 0; c < DH; ++c) s += (hg[c] * invN) * Wout[c * 8 + j];
        out[j] = 1.0f / (1.0f + __expf(-s));
    }
}

extern "C" void kernel_launch(void* const* d_in, const int* in_sizes, int n_in,
                              void* d_out, int out_size, void* d_ws, size_t ws_size,
                              hipStream_t stream) {
    const float* x    = (const float*)d_in[0];
    const float* W0   = (const float*)d_in[1];
    const float* b0   = (const float*)d_in[2];
    const float* a0   = (const float*)d_in[3];
    const float* W1   = (const float*)d_in[4];
    const float* b1   = (const float*)d_in[5];
    const float* a1   = (const float*)d_in[6];
    const float* Wout = (const float*)d_in[7];
    const float* bout = (const float*)d_in[8];
    const int* src0   = (const int*)d_in[9];
    const int* dst0   = (const int*)d_in[10];
    const int* src1   = (const int*)d_in[11];
    const int* dst1   = (const int*)d_in[12];
    float* out        = (float*)d_out;

    // workspace sub-allocation (256-B aligned)
    char* ws = (char*)d_ws;
    size_t off = 0;
    auto alloc = [&](size_t bytes) -> void* {
        void* p = ws + off;
        off = (off + bytes + 255) & ~(size_t)255;
        return p;
    };
    _Float16* xh   = (_Float16*)alloc((size_t)NNODES * DH * sizeof(_Float16));
    _Float16* wt   = (_Float16*)alloc((size_t)2 * DH * DH * sizeof(_Float16));
    float*    wh   = (float*)   alloc((size_t)2 * NNODES * DH * sizeof(float));
    float*    es   = (float*)   alloc((size_t)2 * NNODES * sizeof(float));
    float*    ed   = (float*)   alloc((size_t)2 * NNODES * sizeof(float));
    float*    eval = (float*)   alloc((size_t)2 * NEDGES * sizeof(float));
    unsigned* emaxu= (unsigned*)alloc((size_t)2 * NNODES * sizeof(unsigned));
    float*    esum = (float*)   alloc((size_t)2 * NNODES * sizeof(float));
    float*    agg  = (float*)   alloc((size_t)NNODES * DH * sizeof(float));
    int*      deg  = (int*)     alloc((size_t)NNODES * sizeof(int));
    float*    hg   = (float*)   alloc((size_t)DH * sizeof(float));
    (void)ws_size; (void)in_sizes; (void)n_in; (void)out_size;

    // 1) init accumulators
    init_kernel<<<(NNODES * DH) / 256, 256, 0, stream>>>(agg, emaxu, esum, deg, hg);
    // 2) precision conversion for the WMMA path
    cvt_x<<<(NNODES * DH) / 256, 256, 0, stream>>>(x, xh);
    cvt_w<<<(2 * DH * DH) / 256, 256, 0, stream>>>(W0, W1, wt);
    // 3) wh[rel] = x @ W_rel + b_rel  (v_wmma_f32_16x16x32_f16)
    gemm_wmma<<<(2 * (NNODES / 16) * 8) / 8, 256, 0, stream>>>(xh, wt, b0, b1, wh);
    // 4) per-node attention logits
    esed_kernel<<<(2 * NNODES * 32) / 256, 256, 0, stream>>>(wh, a0, a1, es, ed);
    // 5) softmax over incoming edges + weighted aggregation, per relation
    edge_max<<<NEDGES / 256, 256, 0, stream>>>(src0, dst0, es, ed, eval, emaxu);
    edge_max<<<NEDGES / 256, 256, 0, stream>>>(src1, dst1, es + NNODES, ed + NNODES,
                                               eval + NEDGES, emaxu + NNODES);
    edge_exp<<<NEDGES / 256, 256, 0, stream>>>(dst0, eval, emaxu, esum);
    edge_exp<<<NEDGES / 256, 256, 0, stream>>>(dst1, eval + NEDGES, emaxu + NNODES, esum + NNODES);
    edge_scatter<<<(NEDGES * 32) / 256, 256, 0, stream>>>(src0, dst0, eval, esum, wh, agg);
    edge_scatter<<<(NEDGES * 32) / 256, 256, 0, stream>>>(src1, dst1, eval + NEDGES, esum + NNODES,
                                                          wh + (size_t)NNODES * DH, agg);
    // 6) readout: out-degree histogram + degree-weighted elu reduction
    deg_kernel<<<NEDGES / 256, 256, 0, stream>>>(src0, deg);
    deg_kernel<<<NEDGES / 256, 256, 0, stream>>>(src1, deg);
    readout<<<NNODES / 250, DH, 0, stream>>>(agg, deg, hg);
    // 7) classifier
    final_kernel<<<1, 32, 0, stream>>>(hg, Wout, bout, out);
}
